// MultiScaleRetention_88665304859363
// MI455X (gfx1250) — matrix operand order repaired and checked by reading, not compiled, not run on hardware
//
#include <hip/hip_runtime.h>
#include <hip/hip_bf16.h>

typedef __attribute__((ext_vector_type(16))) __bf16 v16bf;
typedef __attribute__((ext_vector_type(8)))  float  v8f;
typedef __attribute__((ext_vector_type(4)))  float  v4f;
typedef __attribute__((ext_vector_type(4)))  unsigned int v4u;
typedef unsigned short u16;

constexpr int kB  = 2;
constexpr int kS  = 2048;
constexpr int kH  = 1024;
constexpr int kNH = 8;
constexpr int kD  = 128;
constexpr int kBS = kB * kS;   // 4096 tokens

// ---------- helpers ----------
__device__ __forceinline__ u16 f2bf(float f) {
  unsigned u = __builtin_bit_cast(unsigned, f);
  unsigned r = u + 0x7FFFu + ((u >> 16) & 1u);   // round-to-nearest-even
  return (u16)(r >> 16);
}
__device__ __forceinline__ unsigned pack2(float a, float b) {
  return (unsigned)f2bf(a) | ((unsigned)f2bf(b) << 16);
}
__device__ __forceinline__ v8f wmma_bf16(v16bf a, v16bf b, v8f c) {
  return __builtin_amdgcn_wmma_f32_16x16x32_bf16(false, a, false, b, (short)0, c, false, false);
}

struct V4U2 { v4u x, y; };

// A-fragment (16x32 bf16) from row-major MxK LDS tile; two ds_load_b128 per lane
__device__ __forceinline__ v16bf load_a_frag(const u16* p, int ld, int row0, int col0) {
  int lane = threadIdx.x & 31;
  const u16* q = p + (row0 + (lane & 15)) * ld + col0 + ((lane >> 4) << 3);
  V4U2 t;
  t.x = *(const v4u*)q;          // K = base + 0..7
  t.y = *(const v4u*)(q + 16);   // K = base + 16..23
  return __builtin_bit_cast(v16bf, t);
}
// B-fragment (32x16 bf16) where B[k][n] = T[n][k], T row-major NxK; two ds_load_b128
__device__ __forceinline__ v16bf load_bt_frag(const u16* p, int ld, int k0, int n0) {
  int lane = threadIdx.x & 31;
  const u16* q = p + (n0 + (lane & 15)) * ld + k0 + ((lane >> 4) << 3);
  V4U2 t;
  t.x = *(const v4u*)q;          // K = base + 0..7
  t.y = *(const v4u*)(q + 16);   // K = base + 16..23
  return __builtin_bit_cast(v16bf, t);
}

// ---------- kernel 0: weights f32 -> bf16 ----------
__global__ __launch_bounds__(256) void convert_w_kernel(
    const float* __restrict__ Wq, const float* __restrict__ Wk, const float* __restrict__ Wv,
    const float* __restrict__ Wg, const float* __restrict__ Wo,
    u16* __restrict__ Wqb, u16* __restrict__ Wkb, u16* __restrict__ Wvb,
    u16* __restrict__ Wgb, u16* __restrict__ Wob)
{
  int i = blockIdx.x * 256 + threadIdx.x;
  if (i < kNH * kD * kD) { Wqb[i] = f2bf(Wq[i]); Wkb[i] = f2bf(Wk[i]); Wvb[i] = f2bf(Wv[i]); }
  if (i < kH * kH)       { Wgb[i] = f2bf(Wg[i]); Wob[i] = f2bf(Wo[i]); }
}

// ---------- kernel 1: QKV projection + xPos, bf16 out ----------
// grid = 3 * (B*h) * (S/64); block computes a 64x128 tile of Q, K or V
__global__ __launch_bounds__(256) void proj_xpos_kernel(
    const float* __restrict__ X,
    const u16* __restrict__ Wq, const u16* __restrict__ Wk, const u16* __restrict__ Wv,
    u16* __restrict__ Qo, u16* __restrict__ Ko, u16* __restrict__ Vo)
{
  __shared__ __align__(16) u16 Al[64 * 136];     // A row-major MxK
  __shared__ __align__(16) u16 Blt[128 * 136];   // W transposed: [n][k]
  const int tid = threadIdx.x, lane = tid & 31, w = tid >> 5;
  int bx = blockIdx.x;
  const int wm = bx % 3; bx /= 3;
  const int bh = bx & 15;
  const int mt = bx >> 4;
  const int b  = bh >> 3, hh = bh & 7;

  // stage A = Xh[b, rows, hh, :] (f32 -> bf16), 8 elems/thread/iter, b128 both ways
#pragma unroll
  for (int c = 0; c < 4; ++c) {
    int idx8 = tid + 256 * c;                    // 1024 chunks of 8
    int m = idx8 >> 4, k8 = (idx8 & 15) << 3;
    const float* src = X + (size_t)(b * kS + mt * 64 + m) * kH + hh * kD + k8;
    v4f f0 = *(const v4f*)src, f1 = *(const v4f*)(src + 4);
    v4u pk = { pack2(f0[0], f0[1]), pack2(f0[2], f0[3]),
               pack2(f1[0], f1[1]), pack2(f1[2], f1[3]) };
    *(v4u*)&Al[m * 136 + k8] = pk;
  }
  // stage W[h] transposed (coalesced b128 reads, strided b16 stores)
  const u16* W = (wm == 0) ? Wq : (wm == 1) ? Wk : Wv;
#pragma unroll
  for (int c = 0; c < 8; ++c) {
    int idx8 = tid + 256 * c;                    // 2048 chunks of 8
    int k = idx8 >> 4, n8 = (idx8 & 15) << 3;
    v4u wv = *(const v4u*)(W + hh * (kD * kD) + k * kD + n8);
#pragma unroll
    for (int i = 0; i < 4; ++i) {
      Blt[(n8 + 2 * i) * 136 + k]     = (u16)wv[i];
      Blt[(n8 + 2 * i + 1) * 136 + k] = (u16)(wv[i] >> 16);
    }
  }
  __syncthreads();

  const int msub = w & 3, nhalf = w >> 2;
  const int hi = lane >> 4, nlo = lane & 15;
  v8f acc[4] = {};
  for (int kk = 0; kk < 4; ++kk) {
    v16bf a = load_a_frag(Al, 136, msub * 16, kk * 32);
#pragma unroll
    for (int j = 0; j < 4; ++j) {
      v16bf bb = load_bt_frag(Blt, 136, kk * 32, (nhalf * 4 + j) * 16);
      acc[j] = wmma_bf16(a, bb, acc[j]);
    }
  }

  u16* Out = (wm == 0) ? Qo : (wm == 1) ? Ko : Vo;
  const bool isV = (wm == 2), isK = (wm == 1);
#pragma unroll
  for (int j = 0; j < 4; ++j) {
    int e = (nhalf * 4 + j) * 16 + nlo;
    int i2 = e >> 1;
    float sgn = (e & 1) ? 1.f : -1.f;
    float lsc = __log2f((2.f * (float)i2 + 51.2f) * (1.f / 179.2f)); // log2 xpos scale
    float ifr = exp2f(-(float)i2 * (13.2877124f / 64.f));            // 10000^(-i/64)
#pragma unroll
    for (int r = 0; r < 8; ++r) {
      float x = acc[j][r];
      float partner = __shfl_xor(x, 1, 32);      // column e^1, same row
      int s = mt * 64 + msub * 16 + r + 8 * hi;
      float res = x;
      if (!isV) {
        float pos = (float)s;
        float sc = exp2f(lsc * pos * (1.f / 512.f));
        if (isK) sc = 1.f / sc;
        float sn, cs; __sincosf(pos * ifr, &sn, &cs);
        res = x * (cs * sc) + sgn * partner * (sn * sc);
      }
      Out[((size_t)bh * kS + s) * kD + e] = f2bf(res);
    }
  }
}

// ---------- kernel 2: causal retention with decay, flash-style, pipelined ----------
__global__ __launch_bounds__(256) void retention_kernel(
    const u16* __restrict__ Qb, const u16* __restrict__ Kb, const u16* __restrict__ Vb,
    float* __restrict__ Y)
{
  __shared__ __align__(16) u16 Ql[64 * 136];   // Q row-major [m][d]
  __shared__ __align__(16) u16 Kl[64 * 136];   // K row-major [key][d] (B^T for Q.K^T)
  __shared__ __align__(16) u16 Vt[128 * 72];   // V transposed [e][key] (B^T for P.V)
  __shared__ __align__(16) u16 Pl[64 * 72];    // scores row-major [m][key]
  const int tid = threadIdx.x, lane = tid & 31, w = tid >> 5;
  int bx = blockIdx.x;
  const int rt = bx & 31, bh = bx >> 5;
  const int b = bh >> 3, hh = bh & 7;

  const float gamma = 1.f - __expf(-3.4657359f + (-6.2383246f + 3.4657359f) * ((float)hh / 7.f));
  const float l2g = __log2f(gamma);

  const size_t base = (size_t)bh * kS * kD;
  const int pm = (tid + 0) >> 4;               // this thread's staged row (16 chunks/row)
  // thread's 4 staging chunks: idx8 = tid + 256*c  -> m = idx8>>4, k8 = (idx8&15)*8

  // stage Q tile once (b128 both ways)
#pragma unroll
  for (int c = 0; c < 4; ++c) {
    int idx8 = tid + 256 * c;
    int m = idx8 >> 4, k8 = (idx8 & 15) << 3;
    *(v4u*)&Ql[m * 136 + k8] = *(const v4u*)(Qb + base + (size_t)(rt * 64 + m) * kD + k8);
  }
  (void)pm;

  // prefetch K/V tile t=0 into registers
  v4u kreg[4], vreg[4];
#pragma unroll
  for (int c = 0; c < 4; ++c) {
    int idx8 = tid + 256 * c;
    int m = idx8 >> 4, k8 = (idx8 & 15) << 3;
    size_t g = base + (size_t)m * kD + k8;
    kreg[c] = *(const v4u*)(Kb + g);
    vreg[c] = *(const v4u*)(Vb + g);
  }

  const int msub = w >> 1, nh2 = w & 1;
  const int hi = lane >> 4, nlo = lane & 15;
  v8f yacc[4] = {};

  for (int t = 0; t <= rt; ++t) {
    __syncthreads();                 // previous compute done; (t=0: Ql visible after next bar)
    // drain prefetched registers into LDS
#pragma unroll
    for (int c = 0; c < 4; ++c) {
      int idx8 = tid + 256 * c;
      int m = idx8 >> 4, k8 = (idx8 & 15) << 3;
      *(v4u*)&Kl[m * 136 + k8] = kreg[c];
#pragma unroll
      for (int i = 0; i < 4; ++i) {
        Vt[(k8 + 2 * i) * 72 + m]     = (u16)vreg[c][i];
        Vt[(k8 + 2 * i + 1) * 72 + m] = (u16)(vreg[c][i] >> 16);
      }
    }
    // issue next tile's global loads; they overlap the WMMA work below
    if (t < rt) {
#pragma unroll
      for (int c = 0; c < 4; ++c) {
        int idx8 = tid + 256 * c;
        int m = idx8 >> 4, k8 = (idx8 & 15) << 3;
        size_t g = base + (size_t)((t + 1) * 64 + m) * kD + k8;
        kreg[c] = *(const v4u*)(Kb + g);
        vreg[c] = *(const v4u*)(Vb + g);
      }
    }
    __syncthreads();

    // scores S = Q K^T (each wave: 2 of 16 subtiles), decay+mask, -> Pl (bf16)
#pragma unroll
    for (int jj = 0; jj < 2; ++jj) {
      int nsub = nh2 * 2 + jj;
      v8f sacc = {};
#pragma unroll
      for (int kk = 0; kk < 4; ++kk) {
        v16bf a  = load_a_frag(Ql, 136, msub * 16, kk * 32);
        v16bf bb = load_bt_frag(Kl, 136, kk * 32, nsub * 16);
        sacc = wmma_bf16(a, bb, sacc);
      }
#pragma unroll
      for (int r = 0; r < 8; ++r) {
        int mg = rt * 64 + msub * 16 + r + 8 * hi;
        int ng = t * 64 + nsub * 16 + nlo;
        int diff = mg - ng;
        float v = (diff >= 0) ? sacc[r] * exp2f((float)diff * l2g) : 0.f;
        Pl[(msub * 16 + r + 8 * hi) * 72 + nsub * 16 + nlo] = f2bf(v);
      }
    }
    __syncthreads();

    // Y += P @ V (each wave: 4 persistent accumulators)
#pragma unroll
    for (int kk = 0; kk < 2; ++kk) {
      v16bf a = load_a_frag(Pl, 72, msub * 16, kk * 32);
#pragma unroll
      for (int j = 0; j < 4; ++j) {
        v16bf bb = load_bt_frag(Vt, 72, kk * 32, (nh2 * 4 + j) * 16);
        yacc[j] = wmma_bf16(a, bb, yacc[j]);
      }
    }
  }

#pragma unroll
  for (int j = 0; j < 4; ++j) {
    int e = (nh2 * 4 + j) * 16 + nlo;
#pragma unroll
    for (int r = 0; r < 8; ++r) {
      int s = rt * 64 + msub * 16 + r + 8 * hi;
      Y[(size_t)(b * kS + s) * kH + hh * kD + e] = yacc[j][r];
    }
  }
}

// ---------- kernel 3: per-(token, head) GroupNorm, in place ----------
__global__ __launch_bounds__(256) void groupnorm_kernel(
    float* __restrict__ Y, const float* __restrict__ gnw, const float* __restrict__ gnb)
{
  const int lane = threadIdx.x & 31, w = threadIdx.x >> 5;  // w = head
  const size_t base = (size_t)blockIdx.x * kH + w * kD;
  float v[4], s1 = 0.f, s2 = 0.f;
#pragma unroll
  for (int k = 0; k < 4; ++k) { v[k] = Y[base + lane + 32 * k]; s1 += v[k]; s2 += v[k] * v[k]; }
#pragma unroll
  for (int o = 16; o > 0; o >>= 1) { s1 += __shfl_xor(s1, o, 32); s2 += __shfl_xor(s2, o, 32); }
  float mean = s1 * (1.f / 128.f);
  float inv  = rsqrtf(s2 * (1.f / 128.f) - mean * mean + 1e-5f);
#pragma unroll
  for (int k = 0; k < 4; ++k) {
    int e = lane + 32 * k;
    Y[base + e] = (v[k] - mean) * inv * gnw[w * kD + e] + gnb[w * kD + e];
  }
}

// ---------- kernel 4: A2 = silu(X @ W_G) * Yn  (bf16 out) ----------
__global__ __launch_bounds__(256) void gate_gemm_kernel(
    const float* __restrict__ X, const u16* __restrict__ Wgb,
    const float* __restrict__ Yn, u16* __restrict__ A2)
{
  __shared__ __align__(16) u16 Al[64 * 40];    // A chunk row-major 64x32
  __shared__ __align__(16) u16 Blt[128 * 40];  // B chunk transposed [n][k] 128x32
  const int tid = threadIdx.x, lane = tid & 31, w = tid >> 5;
  const int mt = blockIdx.x & 63, nt = blockIdx.x >> 6;
  const int msub = w & 3, nhalf = w >> 2;
  const int hi = lane >> 4, nlo = lane & 15;
  v8f acc[4] = {};
  for (int kk = 0; kk < 32; ++kk) {
    {                                          // A: 2048 elems, 8/thread, f32->bf16
      int m = tid >> 2, k8 = (tid & 3) << 3;
      const float* src = X + (size_t)(mt * 64 + m) * kH + kk * 32 + k8;
      v4f f0 = *(const v4f*)src, f1 = *(const v4f*)(src + 4);
      v4u pk = { pack2(f0[0], f0[1]), pack2(f0[2], f0[3]),
                 pack2(f1[0], f1[1]), pack2(f1[2], f1[3]) };
      *(v4u*)&Al[m * 40 + k8] = pk;
    }
#pragma unroll
    for (int c = 0; c < 2; ++c) {              // B: 4096 elems, 16/thread, transposed store
      int idx8 = tid + 256 * c;
      int k = idx8 >> 4, n8 = (idx8 & 15) << 3;
      v4u wv = *(const v4u*)(Wgb + (size_t)(kk * 32 + k) * kH + nt * 128 + n8);
#pragma unroll
      for (int i = 0; i < 4; ++i) {
        Blt[(n8 + 2 * i) * 40 + k]     = (u16)wv[i];
        Blt[(n8 + 2 * i + 1) * 40 + k] = (u16)(wv[i] >> 16);
      }
    }
    __syncthreads();
    v16bf a = load_a_frag(Al, 40, msub * 16, 0);
#pragma unroll
    for (int j = 0; j < 4; ++j) {
      v16bf bb = load_bt_frag(Blt, 40, 0, (nhalf * 4 + j) * 16);
      acc[j] = wmma_bf16(a, bb, acc[j]);
    }
    __syncthreads();
  }
#pragma unroll
  for (int j = 0; j < 4; ++j)
#pragma unroll
    for (int r = 0; r < 8; ++r) {
      int row = mt * 64 + msub * 16 + r + 8 * hi;
      int col = nt * 128 + (nhalf * 4 + j) * 16 + nlo;
      float g = acc[j][r];
      float val = g * (1.f / (1.f + __expf(-g))) * Yn[(size_t)row * kH + col];
      A2[(size_t)row * kH + col] = f2bf(val);
    }
}

// ---------- kernel 5: out = A2 @ W_O (f32 out) ----------
__global__ __launch_bounds__(256) void out_gemm_kernel(
    const u16* __restrict__ A2, const u16* __restrict__ Wob, float* __restrict__ out)
{
  __shared__ __align__(16) u16 Al[64 * 40];
  __shared__ __align__(16) u16 Blt[128 * 40];
  const int tid = threadIdx.x, lane = tid & 31, w = tid >> 5;
  const int mt = blockIdx.x & 63, nt = blockIdx.x >> 6;
  const int msub = w & 3, nhalf = w >> 2;
  const int hi = lane >> 4, nlo = lane & 15;
  v8f acc[4] = {};
  for (int kk = 0; kk < 32; ++kk) {
    {                                          // A already bf16: b128 both ways
      int m = tid >> 2, k8 = (tid & 3) << 3;
      *(v4u*)&Al[m * 40 + k8] =
          *(const v4u*)(A2 + (size_t)(mt * 64 + m) * kH + kk * 32 + k8);
    }
#pragma unroll
    for (int c = 0; c < 2; ++c) {
      int idx8 = tid + 256 * c;
      int k = idx8 >> 4, n8 = (idx8 & 15) << 3;
      v4u wv = *(const v4u*)(Wob + (size_t)(kk * 32 + k) * kH + nt * 128 + n8);
#pragma unroll
      for (int i = 0; i < 4; ++i) {
        Blt[(n8 + 2 * i) * 40 + k]     = (u16)wv[i];
        Blt[(n8 + 2 * i + 1) * 40 + k] = (u16)(wv[i] >> 16);
      }
    }
    __syncthreads();
    v16bf a = load_a_frag(Al, 40, msub * 16, 0);
#pragma unroll
    for (int j = 0; j < 4; ++j) {
      v16bf bb = load_bt_frag(Blt, 40, 0, (nhalf * 4 + j) * 16);
      acc[j] = wmma_bf16(a, bb, acc[j]);
    }
    __syncthreads();
  }
#pragma unroll
  for (int j = 0; j < 4; ++j)
#pragma unroll
    for (int r = 0; r < 8; ++r) {
      int row = mt * 64 + msub * 16 + r + 8 * hi;
      int col = nt * 128 + (nhalf * 4 + j) * 16 + nlo;
      out[(size_t)row * kH + col] = acc[j][r];
    }
}

extern "C" void kernel_launch(void* const* d_in, const int* in_sizes, int n_in,
                              void* d_out, int out_size, void* d_ws, size_t ws_size,
                              hipStream_t stream) {
  (void)in_sizes; (void)n_in; (void)out_size; (void)ws_size;
  const float* X   = (const float*)d_in[0];
  const float* Wq  = (const float*)d_in[1];
  const float* Wk  = (const float*)d_in[2];
  const float* Wv  = (const float*)d_in[3];
  const float* Wg  = (const float*)d_in[4];
  const float* Wo  = (const float*)d_in[5];
  const float* gnw = (const float*)d_in[6];
  const float* gnb = (const float*)d_in[7];
  float* out = (float*)d_out;

  char* p = (char*)d_ws;
  const size_t NQ = (size_t)kB * kNH * kS * kD;              // 4M elements
  u16*   Qb  = (u16*)p;   p += NQ * 2;
  u16*   Kb  = (u16*)p;   p += NQ * 2;
  u16*   Vb  = (u16*)p;   p += NQ * 2;
  float* Y   = (float*)p; p += (size_t)kBS * kH * 4;
  u16*   A2  = (u16*)p;   p += (size_t)kBS * kH * 2;
  u16*   Wqb = (u16*)p;   p += (size_t)kNH * kD * kD * 2;
  u16*   Wkb = (u16*)p;   p += (size_t)kNH * kD * kD * 2;
  u16*   Wvb = (u16*)p;   p += (size_t)kNH * kD * kD * 2;
  u16*   Wgb = (u16*)p;   p += (size_t)kH * kH * 2;
  u16*   Wob = (u16*)p;   p += (size_t)kH * kH * 2;

  convert_w_kernel<<<(kH * kH + 255) / 256, 256, 0, stream>>>(Wq, Wk, Wv, Wg, Wo,
                                                              Wqb, Wkb, Wvb, Wgb, Wob);
  proj_xpos_kernel<<<3 * kB * kNH * (kS / 64), 256, 0, stream>>>(X, Wqb, Wkb, Wvb, Qb, Kb, Vb);
  retention_kernel<<<kB * kNH * (kS / 64), 256, 0, stream>>>(Qb, Kb, Vb, Y);
  groupnorm_kernel<<<kBS, 256, 0, stream>>>(Y, gnw, gnb);
  gate_gemm_kernel<<<(kBS / 64) * (kH / 128), 256, 0, stream>>>(X, Wgb, Y, A2);
  out_gemm_kernel<<<(kBS / 64) * (kH / 128), 256, 0, stream>>>(A2, Wob, out);
}
